// DeepHandwritingPredictionModel_54855322305217
// MI455X (gfx1250) — matrix operand order repaired and checked by reading, not compiled, not run on hardware
//
#include <hip/hip_runtime.h>
#include <hip/hip_bf16.h>
#include <math.h>

typedef __attribute__((ext_vector_type(16))) __bf16 v16bf;
typedef __attribute__((ext_vector_type(8)))  __bf16 v8bf;
typedef __attribute__((ext_vector_type(8)))  float  v8f;

#define UNITS   400
#define NB      64
#define TSTEPS  1200
#define DIN     3
#define NGATE   1600            // 4*UNITS
#define KPAD    832             // 400 + 400 + 3 padded to 26*32
#define KBLK    26              // KPAD/32
#define NCOL    64              // gate columns per workgroup
#define NWG     25              // 25 * 64 = 1600
#define BLK     256             // 8 waves of 32
#define HN      (NB*UNITS)      // 25600
#define HCHUNK  (HN/8)          // 3200 16-byte chunks per h block
#define MHEAD   121             // 1 + 6*20

// K layout inside A / W (permuted so async 16B chunks are LDS-aligned):
//   k [0,400)   : h_lower   (zero for layer 0)
//   k [400,800) : h_rec
//   k [800,803) : x_t
//   k [803,832) : zero pad

// ---------------------------------------------------------------- utilities

__device__ __forceinline__ float sigmoidf_(float x) {
    return 1.0f / (1.0f + __expf(-x));
}

// Grid-wide barrier among NWG persistent workgroups (atomic generation counter).
__device__ __forceinline__ void grid_barrier(unsigned* cnt, unsigned* gen, unsigned nwg) {
    __syncthreads();
    if (threadIdx.x == 0) {
        __threadfence();
        unsigned g = __hip_atomic_load(gen, __ATOMIC_RELAXED, __HIP_MEMORY_SCOPE_AGENT);
        unsigned arrived = __hip_atomic_fetch_add(cnt, 1u, __ATOMIC_ACQ_REL, __HIP_MEMORY_SCOPE_AGENT);
        if (arrived == nwg - 1u) {
            __hip_atomic_store(cnt, 0u, __ATOMIC_RELAXED, __HIP_MEMORY_SCOPE_AGENT);
            __hip_atomic_fetch_add(gen, 1u, __ATOMIC_RELEASE, __HIP_MEMORY_SCOPE_AGENT);
        } else {
            while (__hip_atomic_load(gen, __ATOMIC_ACQUIRE, __HIP_MEMORY_SCOPE_AGENT) == g) {
                __builtin_amdgcn_s_sleep(1);
            }
        }
        __threadfence();
    }
    __syncthreads();
}

// Async DMA one 16-byte chunk global -> LDS (tracked by ASYNCcnt).
__device__ __forceinline__ void async_copy_b128(unsigned lds_byte_off, const void* gsrc) {
    asm volatile("global_load_async_to_lds_b128 %0, %1, off"
                 :: "v"(lds_byte_off), "v"(gsrc) : "memory");
}

__device__ __forceinline__ void wait_asynccnt0() {
    asm volatile("s_wait_asynccnt 0x0" ::: "memory");
}

// A fragment (16-bit A 16x32, wave32): lanes 0-15 take K {0..7, 16..23},
// lanes 16-31 take K {8..15, 24..31}  (base pointer pre-offset by laneHi*8).
__device__ __forceinline__ v16bf load_a_frag(const __bf16* p) {
    const v8bf* q = (const v8bf*)p;
    v8bf lo = q[0];      // K +0..7
    v8bf hi = q[2];      // K +16..23
    v16bf r;
#pragma unroll
    for (int i = 0; i < 8; ++i) { r[i] = lo[i]; r[i + 8] = hi[i]; }
    return r;
}

// B fragment (16-bit B 32x16, wave32): lane holds column (lane&15),
// lanes 0-15 take K {0..7, 16..23}, lanes 16-31 take K {8..15, 24..31}.
// Weights stored column-major in LDS (base pointer pre-offset by laneHi*8).
__device__ __forceinline__ v16bf load_b_frag(const __bf16* p) {
    const v8bf* q = (const v8bf*)p;
    v8bf lo = q[0];
    v8bf hi = q[2];
    v16bf r;
#pragma unroll
    for (int i = 0; i < 8; ++i) { r[i] = lo[i]; r[i + 8] = hi[i]; }
    return r;
}

// ---------------------------------------------------------------- init

__global__ void lstm_init_kernel(float* Cbuf, __bf16* Hrec, unsigned* bar) {
    int i = blockIdx.x * blockDim.x + threadIdx.x;
    if (i < HN) { Cbuf[i] = 0.0f; Hrec[i] = (__bf16)0.0f; }
    if (i < 2)  { bar[i] = 0u; }
}

// ---------------------------------------------------------------- recurrence
//
// Persistent kernel: NWG workgroups, each owns NCOL gate columns.
// LDS: Wt   [NCOL][KPAD] bf16 (column-major fused [Wx ; Wh] slice)  104 KB
//      Abuf [NB][KPAD]   bf16 ([h_low | h_rec | x_t] rows)          104 KB

__global__ void lstm_layer_kernel(
    const float* __restrict__ inputs,   // (B,T,3) f32
    const float* __restrict__ Wx,       // (wxRows, 1600) f32
    int kkStart,                         // 12 for layer 0 (zero h_low block), else 0
    const float* __restrict__ Wh,       // (400, 1600) f32
    const float* __restrict__ bias,     // (1600,)
    const float* __restrict__ pI,
    const float* __restrict__ pF,
    const float* __restrict__ pO,
    const __bf16* __restrict__ Hlow,    // (T,B,400) bf16 or null (layer 0)
    __bf16* __restrict__ Hout,          // (T,B,400) bf16
    float* __restrict__ Zbuf,           // (64,1600) f32 scratch
    float* __restrict__ Cbuf,           // (64,400) f32
    __bf16* __restrict__ Hrec,          // (64,400) bf16
    unsigned* barCnt, unsigned* barGen)
{
    extern __shared__ __bf16 lds[];
    __bf16* Wt   = lds;                    // NCOL*KPAD
    __bf16* Abuf = lds + NCOL * KPAD;      // NB*KPAD

    const int tid    = threadIdx.x;
    const int wgBase = blockIdx.x * NCOL;
    const bool hasLow = (Hlow != nullptr);

    // LDS byte offset of Abuf (low 32 bits of flat address == wave LDS offset)
    const unsigned abase = (unsigned)(size_t)(const void*)&Abuf[0];

    // ---- one-time: stage fused weight slice (column-major, bf16) ----------
    for (int e = tid; e < NCOL * KPAD; e += BLK) {
        int c = e / KPAD;
        int k = e - c * KPAD;
        int gcol = wgBase + c;
        float w;
        if (k < 400)        w = hasLow ? Wx[(size_t)(k + 3) * NGATE + gcol] : 0.0f;
        else if (k < 800)   w = Wh[(size_t)(k - 400) * NGATE + gcol];
        else if (k < 803)   w = Wx[(size_t)(k - 800) * NGATE + gcol];
        else                w = 0.0f;
        Wt[c * KPAD + k] = (__bf16)w;
    }
    // ---- one-time: zero static regions of Abuf (pad + unused h_low) -------
    for (int e = tid; e < NB * KPAD; e += BLK) {
        int k = e % KPAD;
        bool z = (k >= 803) || (!hasLow && k < 400);
        if (z) Abuf[e] = (__bf16)0.0f;
    }
    __syncthreads();

    const int lane   = tid & 31;
    const int wave   = tid >> 5;
    const int lane16 = lane & 15;
    const int laneHi = lane >> 4;
    const int mt     = wave & 3;            // M tile  (0..3)
    const int nt0    = (wave >> 2) * 2;     // N tiles (pair)

    const int arow  = (mt * 16 + lane16) * KPAD;
    const int bcol0 = ((nt0 + 0) * 16 + lane16) * KPAD;
    const int bcol1 = ((nt0 + 1) * 16 + lane16) * KPAD;
    const int colg0 = wgBase + nt0 * 16 + lane16;

    for (int t = 0; t < TSTEPS; ++t) {
        // ---- stage A_t: async DMA h blocks directly into LDS --------------
        if (hasLow) {
            const __bf16* hl = Hlow + (size_t)t * HN;
            for (int c = tid; c < HCHUNK; c += BLK) {
                int b = c / 50, jj = c - b * 50;                 // 50 chunks/row
                unsigned dst = abase + (unsigned)(b * KPAD + jj * 8) * 2u;
                async_copy_b128(dst, hl + (size_t)c * 8);
            }
            if (t + 1 < TSTEPS) {   // warm GL2 for next step's lower-layer h
                const char* np = (const char*)(Hlow + (size_t)(t + 1) * HN);
                __builtin_prefetch(np + (size_t)tid * 200, 0, 0);
            }
        }
        for (int c = tid; c < HCHUNK; c += BLK) {
            int b = c / 50, jj = c - b * 50;
            unsigned dst = abase + (unsigned)(b * KPAD + 400 + jj * 8) * 2u;
            async_copy_b128(dst, Hrec + (size_t)c * 8);
        }
        if (tid < NB * DIN) {       // tiny x segment stays on the VALU path
            int b = tid / DIN, d = tid - b * DIN;
            Abuf[b * KPAD + 800 + d] = (__bf16)inputs[((size_t)b * TSTEPS + t) * DIN + d];
        }
        wait_asynccnt0();
        __syncthreads();

        // ---- z tile = A_t @ W  via bf16 WMMA, f32 accumulate --------------
        v8f acc0 = {};
        v8f acc1 = {};
#pragma unroll 2
        for (int kk = kkStart; kk < KBLK; ++kk) {
            int ka = kk * 32 + laneHi * 8;
            int kb = kk * 32 + laneHi * 8;
            v16bf af = load_a_frag(&Abuf[arow + ka]);
            v16bf b0 = load_b_frag(&Wt[bcol0 + kb]);
            v16bf b1 = load_b_frag(&Wt[bcol1 + kb]);
            acc0 = __builtin_amdgcn_wmma_f32_16x16x32_bf16(
                false, af, false, b0, (short)0, acc0, false, false);
            acc1 = __builtin_amdgcn_wmma_f32_16x16x32_bf16(
                false, af, false, b1, (short)0, acc1, false, false);
        }
        // D layout: VGPR r -> M = mt*16 + r + 8*laneHi, N = col (lane&15)
#pragma unroll
        for (int r = 0; r < 8; ++r) {
            int row = mt * 16 + r + laneHi * 8;
            Zbuf[row * NGATE + colg0]      = acc0[r];
            Zbuf[row * NGATE + colg0 + 16] = acc1[r];
        }

        grid_barrier(barCnt, barGen, NWG);   // all z columns visible

        // ---- gates / state update (each of 25600 cells) -------------------
        for (int e = blockIdx.x * BLK + tid; e < HN; e += NWG * BLK) {
            int b = e / UNITS, j = e - b * UNITS;
            const float* zr = &Zbuf[b * NGATE];
            float zi = zr[j]             + bias[j];
            float zf = zr[UNITS + j]     + bias[UNITS + j];
            float zg = zr[2 * UNITS + j] + bias[2 * UNITS + j];
            float zo = zr[3 * UNITS + j] + bias[3 * UNITS + j];
            float c  = Cbuf[e];
            float ig = sigmoidf_(zi + pI[j] * c);
            float fg = sigmoidf_(zf + pF[j] * c);
            float cn = fg * c + ig * tanhf(zg);
            float og = sigmoidf_(zo + pO[j] * cn);
            float hn = og * tanhf(cn);
            Cbuf[e] = cn;
            __bf16 hb = (__bf16)hn;
            Hrec[e] = hb;
            Hout[(size_t)t * HN + e] = hb;
        }

        grid_barrier(barCnt, barGen, NWG);   // h_rec ready for next step
    }
}

// ---------------------------------------------------------------- output head
//
// One workgroup per timestep t: LDS-stage Wm (194 KB) + h2 block (51 KB),
// compute Y = h2 @ Wm + bm into LDS, then per-row mixture nonlinearities.

__global__ void mdn_head_kernel(
    const __bf16* __restrict__ H2,      // (T,B,400) bf16
    const float* __restrict__ Wm,       // (400,121)
    const float* __restrict__ bm,       // (121,)
    float* __restrict__ out)            // (B,T,121)
{
    extern __shared__ char smraw[];
    float*  WmS = (float*)smraw;                    // 400*121
    __bf16* hS  = (__bf16*)(WmS + UNITS * MHEAD);   // 64*400
    float*  yS  = (float*)(hS + HN);                // 64*121

    const int t   = blockIdx.x;
    const int tid = threadIdx.x;

    for (int e = tid; e < UNITS * MHEAD; e += BLK) WmS[e] = Wm[e];
    const __bf16* h2r = H2 + (size_t)t * HN;
    for (int e = tid; e < HN; e += BLK) hS[e] = h2r[e];
    __syncthreads();

    for (int e = tid; e < NB * MHEAD; e += BLK) {
        int b = e / MHEAD, c = e - b * MHEAD;
        float acc = bm[c];
        const __bf16* h = &hS[b * UNITS];
#pragma unroll 4
        for (int k = 0; k < UNITS; ++k)
            acc += (float)h[k] * WmS[k * MHEAD + c];
        yS[e] = acc;
    }
    __syncthreads();

    if (tid < NB) {
        const float* y = &yS[tid * MHEAD];
        float* o = &out[((size_t)tid * TSTEPS + t) * MHEAD];
        o[0] = sigmoidf_(y[0]);
        float mx = y[1];
#pragma unroll
        for (int j = 2; j <= 20; ++j) mx = fmaxf(mx, y[j]);
        float ev[20];
        float s = 0.0f;
#pragma unroll
        for (int j = 0; j < 20; ++j) { ev[j] = __expf(y[1 + j] - mx); s += ev[j]; }
        float inv = 1.0f / s;
#pragma unroll
        for (int j = 0; j < 20; ++j) o[1 + j] = ev[j] * inv;
#pragma unroll
        for (int j = 21; j <= 60; ++j)  o[j] = y[j];
#pragma unroll
        for (int j = 61; j <= 100; ++j) o[j] = __expf(y[j]);
#pragma unroll
        for (int j = 101; j <= 120; ++j) o[j] = tanhf(y[j]);
    }
}

// ---------------------------------------------------------------- launch

extern "C" void kernel_launch(void* const* d_in, const int* in_sizes, int n_in,
                              void* d_out, int out_size, void* d_ws, size_t ws_size,
                              hipStream_t stream) {
    const float* inputs = (const float*)d_in[0];
    const float* Wx[3] = { (const float*)d_in[1], (const float*)d_in[7],  (const float*)d_in[13] };
    const float* Wh[3] = { (const float*)d_in[2], (const float*)d_in[8],  (const float*)d_in[14] };
    const float* bb[3] = { (const float*)d_in[3], (const float*)d_in[9],  (const float*)d_in[15] };
    const float* pI[3] = { (const float*)d_in[4], (const float*)d_in[10], (const float*)d_in[16] };
    const float* pF[3] = { (const float*)d_in[5], (const float*)d_in[11], (const float*)d_in[17] };
    const float* pO[3] = { (const float*)d_in[6], (const float*)d_in[12], (const float*)d_in[18] };
    const float* Wm = (const float*)d_in[19];
    const float* bm = (const float*)d_in[20];

    // workspace carve-up (256B aligned)
    char* p = (char*)d_ws;
    size_t off = 0;
    auto take = [&](size_t bytes) -> char* {
        char* r = p + off;
        off = (off + bytes + 255u) & ~(size_t)255u;
        return r;
    };
    float*    Zbuf = (float*)take((size_t)NB * NGATE * sizeof(float));
    float*    Cbuf = (float*)take((size_t)HN * sizeof(float));
    __bf16*   Hrec = (__bf16*)take((size_t)HN * sizeof(__bf16));
    unsigned* bar  = (unsigned*)take(256);
    __bf16*   H[3];
    for (int l = 0; l < 3; ++l)
        H[l] = (__bf16*)take((size_t)TSTEPS * HN * sizeof(__bf16));

    const int lstmSmem = (NCOL + NB) * KPAD * (int)sizeof(__bf16);              // 212992
    const int headSmem = UNITS * MHEAD * 4 + HN * 2 + NB * MHEAD * 4;           // 275776
    (void)hipFuncSetAttribute((const void*)lstm_layer_kernel,
                              hipFuncAttributeMaxDynamicSharedMemorySize, lstmSmem);
    (void)hipFuncSetAttribute((const void*)mdn_head_kernel,
                              hipFuncAttributeMaxDynamicSharedMemorySize, headSmem);

    for (int l = 0; l < 3; ++l) {
        lstm_init_kernel<<<(HN + BLK - 1) / BLK, BLK, 0, stream>>>(Cbuf, Hrec, bar);
        lstm_layer_kernel<<<NWG, BLK, lstmSmem, stream>>>(
            inputs, Wx[l], (l == 0) ? 12 : 0, Wh[l], bb[l],
            pI[l], pF[l], pO[l],
            (l == 0) ? (const __bf16*)nullptr : H[l - 1],
            H[l], Zbuf, Cbuf, Hrec, bar, bar + 1);
    }
    mdn_head_kernel<<<TSTEPS, BLK, headSmem, stream>>>(H[2], Wm, bm, (float*)d_out);
}
